// AttTreeModel_42468636623567
// MI455X (gfx1250) — matrix operand order repaired and checked by reading, not compiled, not run on hardware
//
#include <hip/hip_runtime.h>
#include <math.h>

// ---------------------------------------------------------------------------
// AttTreeModel forward for MI455X (gfx1250): bf16 WMMA GEMMs + fused pointwise
// Weights pre-transposed to [N][K] bf16 so all LDS staging is b128-vectorized.
// ---------------------------------------------------------------------------

typedef __attribute__((ext_vector_type(16))) __bf16 v16bf;
typedef __attribute__((ext_vector_type(8)))  float  v8f;

#define B_   64
#define T_   60
#define L_   196
#define R_   512
#define AH_  512
#define FC_  2048
#define AT_  2048
#define NV_  9488            // V + 1 (logit dim, also embed rows)
#define NVP_ 9600            // logits padded to a multiple of 128

// -------------------------- helper convert kernels -------------------------

// plain fp32 -> bf16 (row-major kept; used for the embedding gather table)
__global__ void k_cvt(const float* __restrict__ s, __bf16* __restrict__ d, long n) {
  long i = blockIdx.x * (long)blockDim.x + threadIdx.x;
  long stride = (long)gridDim.x * blockDim.x;
  for (; i < n; i += stride) d[i] = (__bf16)s[i];
}

// tiled transpose + convert: s[K][N] fp32 -> d[N][ldd] bf16 (d row n, col k)
__global__ __launch_bounds__(256) void k_cvt_t(
    const float* __restrict__ s, __bf16* __restrict__ d, int K, int N, int ldd) {
  __shared__ __bf16 tile[32][33];
  int kb = blockIdx.y * 32, nb = blockIdx.x * 32;
  int tx = threadIdx.x, ty = threadIdx.y;     // block (32, 8)
  for (int yy = ty; yy < 32; yy += 8)
    if (kb + yy < K && nb + tx < N)
      tile[yy][tx] = (__bf16)s[(size_t)(kb + yy) * N + (nb + tx)];
  __syncthreads();
  for (int yy = ty; yy < 32; yy += 8)
    if (nb + yy < N && kb + tx < K)
      d[(size_t)(nb + yy) * ldd + (kb + tx)] = tile[tx][yy];
}

// zero the padded rows [NV_, NVP_) of the transposed logit weight ([N][R_])
__global__ void k_zero_pad_logit(__bf16* __restrict__ d) {
  long n = (long)(NVP_ - NV_) * R_;
  long i = blockIdx.x * (long)blockDim.x + threadIdx.x;
  if (i < n) d[(size_t)NV_ * R_ + i] = (__bf16)0.0f;
}

__global__ void k_pad_bias(const float* __restrict__ s, float* __restrict__ d) {
  int i = blockIdx.x * blockDim.x + threadIdx.x;
  if (i < NVP_) d[i] = (i < NV_) ? s[i] : 0.0f;
}

// ------------------------------ WMMA GEMM ----------------------------------
// C[M,N] = A[M,K] (row major, fp32 or bf16) @ W (bf16, pre-transposed [N][K])
// Block: 256 threads (8 waves). Block tile 64x128; wave tile 16x64 =
// four v_wmma_f32_16x16x32_bf16 sharing one A fragment. Double-buffered LDS,
// one barrier per K-step. M%64==0, N%128==0, K%32==0 guaranteed by caller.
// A/B fragments use contiguous k = 16*kh + s (same in both); fp32 C/D per CDNA5 ISA 7.12.2.

template <bool AF32>
__global__ __launch_bounds__(256) void k_gemm(
    const void* __restrict__ Ain, const __bf16* __restrict__ Bt,
    const float* __restrict__ bias, float* __restrict__ Cf,
    __bf16* __restrict__ Cb, int M, int N, int K, int relu)
{
  __shared__ __bf16 As[2][64][40];    // [m][k] (+8 pad)
  __shared__ __bf16 Bs[2][128][40];   // [n][k] (+8 pad)

  const float*  Af = (const float*)Ain;
  const __bf16* Ab = (const __bf16*)Ain;

  const int tid  = threadIdx.x;
  const int lane = tid & 31;
  const int wave = tid >> 5;
  const int wm   = wave & 3;           // 0..3  (16-row stripe)
  const int wn   = wave >> 2;          // 0..1  (64-col stripe)
  const int m0   = blockIdx.y * 64;
  const int n0   = blockIdx.x * 128;

  const int kh  = lane >> 4;           // lane half (ISA K-split)
  const int l16 = lane & 15;

  // staging assignments (contiguous 16B/32B chunks -> b128 LDS traffic)
  const int ar  = tid >> 2, ac  = (tid & 3) * 8;    // A: 64 rows x 4 x 8
  const int bn  = tid >> 1, bkg = (tid & 1) * 16;   // B: 128 rows x 2 x 16

  v8f acc0 = {};
  v8f acc1 = {};
  v8f acc2 = {};
  v8f acc3 = {};

  auto stage = [&](int buf, int k0) {
    if (AF32) {
      const float* src = Af + (size_t)(m0 + ar) * K + (k0 + ac);
#pragma unroll
      for (int q = 0; q < 8; ++q) As[buf][ar][ac + q] = (__bf16)src[q];
    } else {
      const __bf16* src = Ab + (size_t)(m0 + ar) * K + (k0 + ac);
#pragma unroll
      for (int q = 0; q < 8; ++q) As[buf][ar][ac + q] = src[q];
    }
    const __bf16* srcB = Bt + (size_t)(n0 + bn) * K + (k0 + bkg);
    if (k0 + 32 < K) __builtin_prefetch(srcB + 32, 0, 1);
#pragma unroll
    for (int q = 0; q < 16; ++q) Bs[buf][bn][bkg + q] = srcB[q];
  };

  stage(0, 0);
  __syncthreads();

  int cur = 0;
  for (int k0 = 0; k0 < K; k0 += 32) {
    int nxt = cur ^ 1;
    if (k0 + 32 < K) stage(nxt, k0 + 32);

    // ---- A fragment: two contiguous 8-element chunks per lane ----
    v16bf a;
    const __bf16* Ar = &As[cur][wm * 16 + l16][0];
#pragma unroll
    for (int q = 0; q < 8; ++q) {
      a[q]     = Ar[16 * kh + q];           // K = 16*kh + q
      a[8 + q] = Ar[16 * kh + 8 + q];       // K = 16*kh + 8 + q
    }
    // ---- B fragments: 16 contiguous K values per lane per tile ----
    v16bf b0, b1, b2, b3;
    const __bf16* Br0 = &Bs[cur][wn * 64 + l16][16 * kh];
    const __bf16* Br1 = &Bs[cur][wn * 64 + 16 + l16][16 * kh];
    const __bf16* Br2 = &Bs[cur][wn * 64 + 32 + l16][16 * kh];
    const __bf16* Br3 = &Bs[cur][wn * 64 + 48 + l16][16 * kh];
#pragma unroll
    for (int q = 0; q < 16; ++q) {
      b0[q] = Br0[q];
      b1[q] = Br1[q];
      b2[q] = Br2[q];
      b3[q] = Br3[q];
    }

    acc0 = __builtin_amdgcn_wmma_f32_16x16x32_bf16(false, a, false, b0,
                                                   (short)0, acc0, false, false);
    acc1 = __builtin_amdgcn_wmma_f32_16x16x32_bf16(false, a, false, b1,
                                                   (short)0, acc1, false, false);
    acc2 = __builtin_amdgcn_wmma_f32_16x16x32_bf16(false, a, false, b2,
                                                   (short)0, acc2, false, false);
    acc3 = __builtin_amdgcn_wmma_f32_16x16x32_bf16(false, a, false, b3,
                                                   (short)0, acc3, false, false);
    __syncthreads();
    cur = nxt;
  }

  // ---- epilogue: C layout lanes 0-15 -> M=r, lanes 16-31 -> M=8+r ----
  const int mb = kh * 8;
#pragma unroll
  for (int t = 0; t < 4; ++t) {
    v8f acc = (t == 0) ? acc0 : (t == 1) ? acc1 : (t == 2) ? acc2 : acc3;
    int c = n0 + wn * 64 + t * 16 + l16;
#pragma unroll
    for (int r = 0; r < 8; ++r) {
      int gm = m0 + wm * 16 + mb + r;
      float v = acc[r];
      if (bias) v += bias[c];
      if (relu) v = fmaxf(v, 0.0f);
      if (Cf) Cf[(size_t)gm * N + c] = v;
      if (Cb) Cb[(size_t)gm * N + c] = (__bf16)v;
    }
  }
}

// --------------------------- per-step kernels ------------------------------

// h_prev = p_h + s_h, c_prev = p_c + s_c ; also bf16 h_prev into x' slot 3
__global__ void k_build_state(const int* __restrict__ parent_idx,
                              const float* __restrict__ hs,
                              const float* __restrict__ cs,
                              float* __restrict__ hprev, float* __restrict__ cprev,
                              __bf16* __restrict__ xprime, int i)
{
  int idx = blockIdx.x * blockDim.x + threadIdx.x;
  if (idx >= B_ * R_) return;
  int b = idx / R_, r = idx % R_;
  float h = 0.0f, c = 0.0f;
  if (i > 0) {
    int pidx = parent_idx[b * T_ + i];
    h = hs[((size_t)b * T_ + pidx) * R_ + r];
    c = cs[((size_t)b * T_ + pidx) * R_ + r];
    if ((i % 3) != 1) {               // not first child: add sibling state
      h += hs[((size_t)b * T_ + (i - 1)) * R_ + r];
      c += cs[((size_t)b * T_ + (i - 1)) * R_ + r];
    }
  }
  hprev[idx] = h;
  cprev[idx] = c;
  xprime[(size_t)b * 2048 + 1536 + r] = (__bf16)h;
}

// x' slots 0/1: p_xt = relu(embed[parent token]), s_xt = relu(embed[sibling])
__global__ void k_build_x(const int* __restrict__ seqtree,
                          const int* __restrict__ parent_idx,
                          const __bf16* __restrict__ embed,
                          const __bf16* __restrict__ pfc,
                          __bf16* __restrict__ xprime, int i)
{
  int idx = blockIdx.x * blockDim.x + threadIdx.x;
  if (idx >= B_ * R_) return;
  int b = idx / R_, r = idx % R_;
  __bf16 px, sx;
  if (i == 0) {
    px = pfc[idx];                     // already relu'd fc projection
    sx = (__bf16)0.0f;
  } else {
    int pidx = parent_idx[b * T_ + i];
    int p_it = seqtree[b * T_ + pidx];
    px = (__bf16)fmaxf((float)embed[(size_t)p_it * R_ + r], 0.0f);
    if ((i % 3) == 1) {
      sx = (__bf16)0.0f;
    } else {
      int s_it = seqtree[b * T_ + (i - 1)];
      sx = (__bf16)fmaxf((float)embed[(size_t)s_it * R_ + r], 0.0f);
    }
  }
  xprime[(size_t)b * 2048 + r]       = px;
  xprime[(size_t)b * 2048 + 512 + r] = sx;
}

// scores[b,l] = sum_k tanh(pp_att[b,l,k] + ha[b,k]) * alpha[k]
__global__ __launch_bounds__(128) void k_attn_score(
    const float* __restrict__ pp_att, const float* __restrict__ ha,
    const float* __restrict__ alpha, float* __restrict__ scores)
{
  __shared__ float red[128];
  int bl = blockIdx.x;                 // 0 .. B*L-1
  int b  = bl / L_;
  int tid = threadIdx.x;
  float acc = 0.0f;
  for (int k = tid; k < AH_; k += 128)
    acc += tanhf(pp_att[(size_t)bl * AH_ + k] + ha[(size_t)b * AH_ + k]) * alpha[k];
  red[tid] = acc;
  __syncthreads();
  for (int s = 64; s > 0; s >>= 1) {
    if (tid < s) red[tid] += red[tid + s];
    __syncthreads();
  }
  if (tid == 0) scores[bl] = red[0];
}

// masked softmax over L, then ctx[b,r] = sum_l w[l] * p_att[b,l,r] into x' slot 2
__global__ __launch_bounds__(256) void k_attn_ctx(
    const float* __restrict__ scores, const float* __restrict__ mask,
    const __bf16* __restrict__ p_att, __bf16* __restrict__ xprime)
{
  __shared__ float red[256];
  __shared__ float w[L_];
  int b = blockIdx.x, tid = threadIdx.x;
  float s = -1e30f;
  if (tid < L_) {
    float v = scores[b * L_ + tid];
    s = (mask[b * L_ + tid] > 0.0f) ? v : -1e9f;
    w[tid] = s;
  }
  red[tid] = s;
  __syncthreads();
  for (int st = 128; st > 0; st >>= 1) {
    if (tid < st) red[tid] = fmaxf(red[tid], red[tid + st]);
    __syncthreads();
  }
  float mx = red[0];
  __syncthreads();
  float e = 0.0f;
  if (tid < L_) { e = expf(w[tid] - mx); w[tid] = e; }
  red[tid] = e;
  __syncthreads();
  for (int st = 128; st > 0; st >>= 1) {
    if (tid < st) red[tid] += red[tid + st];
    __syncthreads();
  }
  float inv = 1.0f / red[0];
  __syncthreads();
  if (tid < L_) w[tid] *= inv;
  __syncthreads();
  for (int r = tid; r < R_; r += 256) {
    float acc = 0.0f;
    for (int l = 0; l < L_; ++l)
      acc += w[l] * (float)p_att[((size_t)b * L_ + l) * R_ + r];
    xprime[(size_t)b * 2048 + 1024 + r] = (__bf16)acc;
  }
}

// LSTM cell pointwise: gates layout [i, f, o, g] each of width R
__global__ void k_lstm_cell(const float* __restrict__ gates,
                            const float* __restrict__ cprev,
                            float* __restrict__ hs, float* __restrict__ cs,
                            __bf16* __restrict__ hcur, int i)
{
  int idx = blockIdx.x * blockDim.x + threadIdx.x;
  if (idx >= B_ * R_) return;
  int b = idx / R_, r = idx % R_;
  const float* g = gates + (size_t)b * 2048;
  float ig = g[r], fg = g[512 + r], og = g[1024 + r], gg = g[1536 + r];
  float si = 1.0f / (1.0f + expf(-ig));
  float sf = 1.0f / (1.0f + expf(-fg));
  float so = 1.0f / (1.0f + expf(-og));
  float c = sf * cprev[idx] + si * tanhf(gg);
  float h = so * tanhf(c);
  hs[((size_t)b * T_ + i) * R_ + r] = h;
  cs[((size_t)b * T_ + i) * R_ + r] = c;
  hcur[idx] = (__bf16)h;
}

// log_softmax over first NV_ entries of padded logits -> d_out[b, i, :]
__global__ __launch_bounds__(256) void k_logsoftmax(
    const float* __restrict__ logits, float* __restrict__ out, int i)
{
  __shared__ float red[256];
  int b = blockIdx.x, tid = threadIdx.x;
  const float* lg = logits + (size_t)b * NVP_;
  float mx = -1e30f;
  for (int v = tid; v < NV_; v += 256) mx = fmaxf(mx, lg[v]);
  red[tid] = mx;
  __syncthreads();
  for (int s = 128; s > 0; s >>= 1) {
    if (tid < s) red[tid] = fmaxf(red[tid], red[tid + s]);
    __syncthreads();
  }
  mx = red[0];
  __syncthreads();
  float sum = 0.0f;
  for (int v = tid; v < NV_; v += 256) sum += expf(lg[v] - mx);
  red[tid] = sum;
  __syncthreads();
  for (int s = 128; s > 0; s >>= 1) {
    if (tid < s) red[tid] += red[tid + s];
    __syncthreads();
  }
  float lse = mx + logf(red[0]);
  float* o = out + ((size_t)b * T_ + i) * NV_;
  for (int v = tid; v < NV_; v += 256) o[v] = lg[v] - lse;
}

// ------------------------------- launcher ----------------------------------

extern "C" void kernel_launch(void* const* d_in, const int* in_sizes, int n_in,
                              void* d_out, int out_size, void* d_ws, size_t ws_size,
                              hipStream_t stream)
{
  (void)in_sizes; (void)n_in; (void)out_size; (void)ws_size;

  const float* fc_feats  = (const float*)d_in[0];
  const float* att_feats = (const float*)d_in[1];
  const float* att_masks = (const float*)d_in[3];
  const int*   seqtree   = (const int*)d_in[4];
  const int*   parent    = (const int*)d_in[5];
  const float* embed_W   = (const float*)d_in[7];
  const float* fc_W      = (const float*)d_in[8];
  const float* fc_b      = (const float*)d_in[9];
  const float* att_W     = (const float*)d_in[10];
  const float* att_b     = (const float*)d_in[11];
  const float* ctx2att_W = (const float*)d_in[12];
  const float* ctx2att_b = (const float*)d_in[13];
  const float* h2att_W   = (const float*)d_in[14];
  const float* h2att_b   = (const float*)d_in[15];
  const float* alpha_w   = (const float*)d_in[16];
  const float* lstm_Wx   = (const float*)d_in[17];
  const float* lstm_Wh   = (const float*)d_in[18];
  const float* lstm_b    = (const float*)d_in[19];
  const float* logit_W   = (const float*)d_in[20];
  const float* logit_b   = (const float*)d_in[21];
  float* out = (float*)d_out;

  char* ws = (char*)d_ws;
  size_t off = 0;
  auto alloc = [&](size_t bytes) -> void* {
    off = (off + 255) & ~(size_t)255;
    void* p = ws + off;
    off += bytes;
    return p;
  };

  const int ML = B_ * L_;   // 12544 rows for att precompute

  // transposed bf16 weights: Wt[n][k]
  __bf16* embed_bf  = (__bf16*)alloc((size_t)NV_ * R_ * 2);      // row-major (gather)
  __bf16* fcWt      = (__bf16*)alloc((size_t)R_ * FC_ * 2);
  __bf16* attWt     = (__bf16*)alloc((size_t)R_ * AT_ * 2);
  __bf16* c2aWt     = (__bf16*)alloc((size_t)AH_ * R_ * 2);
  __bf16* h2aWt     = (__bf16*)alloc((size_t)AH_ * R_ * 2);
  __bf16* WlstmT    = (__bf16*)alloc((size_t)2048 * 2048 * 2);   // [n=2048][k=2048]
  __bf16* logitWt   = (__bf16*)alloc((size_t)NVP_ * R_ * 2);     // [n=NVP_][k=512]
  float*  logitb_p  = (float*) alloc((size_t)NVP_ * 4);
  __bf16* pfc_bf    = (__bf16*)alloc((size_t)B_ * R_ * 2);
  __bf16* patt_bf   = (__bf16*)alloc((size_t)ML * R_ * 2);
  float*  ppatt     = (float*) alloc((size_t)ML * AH_ * 4);
  float*  hs        = (float*) alloc((size_t)B_ * T_ * R_ * 4);
  float*  cs        = (float*) alloc((size_t)B_ * T_ * R_ * 4);
  float*  hprev     = (float*) alloc((size_t)B_ * R_ * 4);
  float*  cprev     = (float*) alloc((size_t)B_ * R_ * 4);
  __bf16* xprime    = (__bf16*)alloc((size_t)B_ * 2048 * 2);     // [p_xt|s_xt|ctx|h]
  float*  ha        = (float*) alloc((size_t)B_ * AH_ * 4);
  float*  scores    = (float*) alloc((size_t)B_ * L_ * 4);
  float*  gates     = (float*) alloc((size_t)B_ * 2048 * 4);
  __bf16* hcur      = (__bf16*)alloc((size_t)B_ * R_ * 2);
  float*  logits    = (float*) alloc((size_t)B_ * NVP_ * 4);

  auto cvt_t = [&](const float* s, __bf16* d, int K, int N, int ldd) {
    dim3 g((N + 31) / 32, (K + 31) / 32), blk(32, 8);
    k_cvt_t<<<g, blk, 0, stream>>>(s, d, K, N, ldd);
  };
  auto gemm = [&](const void* A, bool af32, const __bf16* Bt, const float* bias,
                  float* Cf, __bf16* Cb, int M, int N, int K, int relu) {
    dim3 g(N / 128, M / 64), blk(256);
    if (af32) k_gemm<true ><<<g, blk, 0, stream>>>(A, Bt, bias, Cf, Cb, M, N, K, relu);
    else      k_gemm<false><<<g, blk, 0, stream>>>(A, Bt, bias, Cf, Cb, M, N, K, relu);
  };

  // ---- weight conversion / transposition (re-done every call) ----
  k_cvt<<<2048, 256, 0, stream>>>(embed_W, embed_bf, (long)NV_ * R_);
  cvt_t(fc_W,      fcWt,  FC_, R_,  FC_);
  cvt_t(att_W,     attWt, AT_, R_,  AT_);
  cvt_t(ctx2att_W, c2aWt, R_,  AH_, R_);
  cvt_t(h2att_W,   h2aWt, R_,  AH_, R_);
  cvt_t(lstm_Wx,   WlstmT,        1536, 2048, 2048);   // combined k rows 0..1535
  cvt_t(lstm_Wh,   WlstmT + 1536, 512,  2048, 2048);   // combined k rows 1536..2047
  cvt_t(logit_W,   logitWt, R_, NV_, R_);
  k_zero_pad_logit<<<((NVP_ - NV_) * R_ + 255) / 256, 256, 0, stream>>>(logitWt);
  k_pad_bias<<<(NVP_ + 255) / 256, 256, 0, stream>>>(logit_b, logitb_p);

  // ---- feature precompute (big WMMA GEMMs) ----
  gemm(fc_feats,  true,  fcWt,  fc_b,      nullptr, pfc_bf,  B_, R_,  FC_, 1);
  gemm(att_feats, true,  attWt, att_b,     nullptr, patt_bf, ML, R_,  AT_, 1);
  gemm(patt_bf,   false, c2aWt, ctx2att_b, ppatt,   nullptr, ML, AH_, R_,  0);

  // ---- sequential tree scan ----
  for (int i = 0; i < T_; ++i) {
    k_build_state<<<128, 256, 0, stream>>>(parent, hs, cs, hprev, cprev, xprime, i);
    gemm(hprev, true, h2aWt, h2att_b, ha, nullptr, B_, AH_, R_, 0);
    k_attn_score<<<B_ * L_, 128, 0, stream>>>(ppatt, ha, alpha_w, scores);
    k_attn_ctx<<<B_, 256, 0, stream>>>(scores, att_masks, patt_bf, xprime);
    k_build_x<<<128, 256, 0, stream>>>(seqtree, parent, embed_bf, pfc_bf, xprime, i);
    gemm(xprime, false, WlstmT, lstm_b, gates, nullptr, B_, 2048, 2048, 0);
    k_lstm_cell<<<128, 256, 0, stream>>>(gates, cprev, hs, cs, hcur, i);
    gemm(hcur, false, logitWt, logitb_p, logits, nullptr, B_, NVP_, R_, 0);
    k_logsoftmax<<<B_, 256, 0, stream>>>(logits, out, i);
  }
}
